// TransformerEncoderLayer_4123168604645
// MI455X (gfx1250) — matrix operand order repaired and checked by reading, not compiled, not run on hardware
//
#include <hip/hip_runtime.h>
#include <math.h>

// ---------------------------------------------------------------------------
// MI455X (gfx1250, wave32) implementation of the sampled-attention encoder
// layer. All large matmuls run on v_wmma_f32_16x16x32_bf16 (f32 accumulate).
// Attention probabilities are never materialized (flash-style 2-pass).
// ---------------------------------------------------------------------------

typedef __attribute__((ext_vector_type(16))) __bf16 v16bf;
typedef __attribute__((ext_vector_type(8)))  __bf16 v8bf;
typedef __attribute__((ext_vector_type(8)))  float  v8f;

#define BATCH 32
#define TT    512
#define DD    256
#define LL    256
#define QHEADS 8      // H_MIX, dh = 32
#define AHEADS 4      // H_ATT, qd = 64

// ---------------- workspace layout (bytes) ----------------
constexpr size_t WS_MIXW  = 0;                                  // 768x256 bf16 (q rows pre-scaled by 1/sqrt(32))
constexpr size_t WS_OUTW  = WS_MIXW  + 768ull*256*2;            // 256x256 bf16
constexpr size_t WS_FF1W  = WS_OUTW  + 256ull*256*2;            // 512x256 bf16
constexpr size_t WS_FF2W  = WS_FF1W  + 512ull*256*2;            // 256x512 bf16
constexpr size_t WS_ATTW  = WS_FF2W  + 256ull*512*2;            // 768x256 bf16
constexpr size_t WS_MIXB  = WS_ATTW  + 768ull*256*2;            // 768 f32 (scaled qkv bias)
constexpr size_t WS_XBF   = WS_MIXB  + 4096;                    // 16384x256 bf16  (later reused as O_bf 8192x256)
constexpr size_t WS_QKV   = WS_XBF   + 16384ull*256*2;          // 16384x768 bf16
constexpr size_t WS_VT    = WS_QKV   + 16384ull*768*2;          // [b][h][32][512] bf16
constexpr size_t WS_RMAX  = WS_VT    + 32ull*8*32*512*2;        // [b][h][512] f32
constexpr size_t WS_RINV  = WS_RMAX  + 32ull*8*512*4;
constexpr size_t WS_WGT   = WS_RINV  + 32ull*8*512*4;           // [b][512] f32
constexpr size_t WS_SAMP  = WS_WGT   + 32ull*512*4;             // [b][256] int
constexpr size_t WS_HPRE  = WS_SAMP  + 32ull*256*4;             // 8192x256 f32 (also reused as ffn2 output)
constexpr size_t WS_HM    = WS_HPRE  + 8192ull*256*4;           // 8192x256 f32
constexpr size_t WS_KV    = WS_HM    + 8192ull*256*4;           // 8192x512 f32
constexpr size_t WS_END   = WS_KV    + 8192ull*512*4;
// aliases carved out of WS_QKV once pass2 is done with qkv:
constexpr size_t WS_HMBF  = WS_QKV;                             // 8192x256 bf16
constexpr size_t WS_FF1O  = WS_QKV + 8192ull*256*2;             // 8192x512 bf16
constexpr size_t WS_INTRB = WS_FF1O + 8192ull*512*2;            // 8192x256 bf16

// output element offsets (f32-sized slots)
constexpr size_t OUT_GROUPED = 32ull*257*256;                   // 2105344
constexpr size_t OUT_IDX     = OUT_GROUPED + 8ull*4*256;        // 2113536

// ---------------- WMMA helpers ----------------
__device__ __forceinline__ v16bf load_a16(const __bf16* lo, const __bf16* hi) {
  v8bf a = *(const v8bf*)lo;
  v8bf b = *(const v8bf*)hi;
  v16bf r;
#pragma unroll
  for (int i = 0; i < 8; ++i) { r[i] = a[i]; r[i + 8] = b[i]; }
  return r;
}

__device__ __forceinline__ v8f wmma_bf16(v16bf a, v16bf b, v8f c) {
  return __builtin_amdgcn_wmma_f32_16x16x32_bf16(false, a, false, b, (short)0, c, false, false);
}

__device__ __forceinline__ float blk_sum256(float v, float* red) {
  int t = threadIdx.x;
  red[t] = v; __syncthreads();
#pragma unroll
  for (int s = 128; s > 0; s >>= 1) {
    if (t < s) red[t] += red[t + s];
    __syncthreads();
  }
  float r = red[0]; __syncthreads();
  return r;
}

// ---------------- conversion / setup kernels ----------------
__global__ void k_cvt_scale(const float* __restrict__ s, __bf16* __restrict__ d, int n, float scale) {
  int i = blockIdx.x * 256 + threadIdx.x;
  if (i < n) d[i] = (__bf16)(s[i] * scale);
}

__global__ void k_scale_qbias(const float* __restrict__ b, float* __restrict__ o) {
  int i = blockIdx.x * 256 + threadIdx.x;
  if (i < 768) o[i] = (i < 256) ? b[i] * 0.17677669529663687f : b[i]; // 1/sqrt(32)
}

__global__ void k_cvt_x(const float* __restrict__ src, __bf16* __restrict__ x) {
  int i = blockIdx.x * 256 + threadIdx.x;     // 16384*256 total
  int d = i & 255, t = (i >> 8) & 511, b = i >> 17;
  x[i] = (__bf16)src[((size_t)(b * 513 + 1 + t)) * 256 + d];
}

__global__ void k_zero_f32(float* __restrict__ p, int n) {
  int i = blockIdx.x * 256 + threadIdx.x;
  if (i < n) p[i] = 0.f;
}

__global__ void k_make_vT(const __bf16* __restrict__ qkv, __bf16* __restrict__ vT) {
  int i = blockIdx.x * 256 + threadIdx.x;     // 32*8*32*512 total
  int t = i & 511, dd = (i >> 9) & 31, h = (i >> 14) & 7, b = i >> 17;
  vT[i] = qkv[(size_t)(b * 512 + t) * 768 + 512 + h * 32 + dd];
}

// ---------------- generic bf16 GEMM: out = A(MxK) * W(NxK)^T + bias ----------------
// one wave per 16(M) x 64(N) tile; flags: bit0 = relu, bit1 = bf16 output
__global__ void k_gemm_bf16(const __bf16* __restrict__ A, const __bf16* __restrict__ W,
                            const float* __restrict__ bias, void* __restrict__ out,
                            int M, int N, int K, int flags) {
  const int l  = threadIdx.x;
  const int lo = l & 15, kh = l >> 4;
  const int m0 = blockIdx.x * 16;
  const int n0 = blockIdx.y * 64;
  const __bf16* arow = A + (size_t)(m0 + lo) * K;
  v8f acc[4] = {};
  for (int k0 = 0; k0 < K; k0 += 32) {
    v16bf a = load_a16(arow + k0 + kh * 16, arow + k0 + kh * 16 + 8);
#pragma unroll
    for (int t = 0; t < 4; ++t) {
      const __bf16* bp = W + (size_t)(n0 + t * 16 + lo) * K + k0 + kh * 16;
      v16bf b = *(const v16bf*)bp;
      acc[t] = wmma_bf16(a, b, acc[t]);
    }
  }
#pragma unroll
  for (int t = 0; t < 4; ++t) {
    int n = n0 + t * 16 + lo;
    float bs = bias[n];
#pragma unroll
    for (int v = 0; v < 8; ++v) {
      int m = m0 + v + 8 * kh;
      float x = acc[t][v] + bs;
      if (flags & 1) x = fmaxf(x, 0.f);
      if (flags & 2) ((__bf16*)out)[(size_t)m * N + n] = (__bf16)x;
      else           ((float*)out)[(size_t)m * N + n]  = x;
    }
  }
}

// ---------------- attention pass 1: softmax stats + weight[b][k] ----------------
// grid = 32 batches * 32 qtiles; block = 256 (8 waves, wave == head in stage 1)
__global__ void k_attn_pass1(const __bf16* __restrict__ qkv, float* __restrict__ rmax,
                             float* __restrict__ rinv, float* __restrict__ wgt) {
  __shared__ float s_max[8][16];
  __shared__ float s_inv[8][16];
  const int l   = threadIdx.x & 31;
  const int wid = threadIdx.x >> 5;
  const int lo  = l & 15, kh = l >> 4;
  const int b   = blockIdx.x >> 5;
  const int qt  = blockIdx.x & 31;

  { // stage 1: online softmax stats for head h = wid over all 512 keys
    const int h = wid;
    const __bf16* qbase = qkv + (size_t)(b * 512 + qt * 16 + lo) * 768 + h * 32;
    v16bf qa = load_a16(qbase + kh * 16, qbase + kh * 16 + 8);
    float M[8], S[8];
#pragma unroll
    for (int v = 0; v < 8; ++v) { M[v] = -3.0e38f; S[v] = 0.f; }
    for (int kt = 0; kt < 32; ++kt) {
      const __bf16* kbase = qkv + (size_t)(b * 512 + kt * 16 + lo) * 768 + 256 + h * 32 + kh * 16;
      v16bf kb = *(const v16bf*)kbase;
      v8f s = {};
      s = wmma_bf16(qa, kb, s);
#pragma unroll
      for (int v = 0; v < 8; ++v) {
        float sv = s[v];
        float tm = sv;                         // row-max across the 16 lanes of this half
        tm = fmaxf(tm, __shfl_xor(tm, 1));
        tm = fmaxf(tm, __shfl_xor(tm, 2));
        tm = fmaxf(tm, __shfl_xor(tm, 4));
        tm = fmaxf(tm, __shfl_xor(tm, 8));
        float nm = fmaxf(M[v], tm);
        float e  = __expf(sv - nm);
        float es = e;
        es += __shfl_xor(es, 1);
        es += __shfl_xor(es, 2);
        es += __shfl_xor(es, 4);
        es += __shfl_xor(es, 8);
        S[v] = S[v] * __expf(M[v] - nm) + es;
        M[v] = nm;
      }
    }
    if (lo == 0) {
#pragma unroll
      for (int v = 0; v < 8; ++v) {
        int m = v + 8 * kh;
        float inv = 1.f / S[v];
        s_max[h][m] = M[v];
        s_inv[h][m] = inv;
        size_t o = (size_t)(b * 8 + h) * 512 + qt * 16 + m;
        rmax[o] = M[v];
        rinv[o] = inv;
      }
    }
  }
  __syncthreads();

  // stage 2: weight[b][k] += sum_q max_h A[b,h,q,k]; each wave owns 4 k-tiles
  for (int kk = 0; kk < 4; ++kk) {
    int kt = wid * 4 + kk;
    float amax[8];
#pragma unroll
    for (int v = 0; v < 8; ++v) amax[v] = 0.f;
    for (int h = 0; h < 8; ++h) {
      const __bf16* qbase = qkv + (size_t)(b * 512 + qt * 16 + lo) * 768 + h * 32;
      v16bf qa = load_a16(qbase + kh * 16, qbase + kh * 16 + 8);
      const __bf16* kbase = qkv + (size_t)(b * 512 + kt * 16 + lo) * 768 + 256 + h * 32 + kh * 16;
      v16bf kb = *(const v16bf*)kbase;
      v8f s = {};
      s = wmma_bf16(qa, kb, s);
#pragma unroll
      for (int v = 0; v < 8; ++v) {
        int m = v + 8 * kh;
        float a = __expf(s[v] - s_max[h][m]) * s_inv[h][m];
        amax[v] = fmaxf(amax[v], a);
      }
    }
    float p = 0.f;
#pragma unroll
    for (int v = 0; v < 8; ++v) p += amax[v];
    atomicAdd(&wgt[b * 512 + kt * 16 + lo], p);
  }
}

// ---------------- stable descending top-k via rank ----------------
__global__ void k_topk_rank(const float* __restrict__ wgt, int* __restrict__ sampled) {
  __shared__ float w[512];
  int b = blockIdx.x, t = threadIdx.x;
  w[t] = wgt[b * 512 + t];
  __syncthreads();
  float mv = w[t];
  int rank = 0;
  for (int j = 0; j < 512; ++j) {
    float wj = w[j];
    rank += (wj > mv) || (wj == mv && j < t);
  }
  if (rank < 256) sampled[b * 256 + rank] = t;
}

// ---------------- attention pass 2: O for selected rows ----------------
// grid = 32 batches * 16 ltiles; block = 128 (4 waves, each handles 2 heads)
__global__ void k_attn_pass2(const __bf16* __restrict__ qkv, const __bf16* __restrict__ vT,
                             const float* __restrict__ rmax, const float* __restrict__ rinv,
                             const int* __restrict__ sampled, __bf16* __restrict__ Obf) {
  __shared__ __bf16 P[4 * 16 * 512];            // 64 KB, one 16x512 tile per wave
  const int l   = threadIdx.x & 31;
  const int wid = threadIdx.x >> 5;             // 0..3
  const int lo  = l & 15, kh = l >> 4;
  const int b   = blockIdx.x >> 4;
  const int lt  = blockIdx.x & 15;
  __bf16* Pw = P + wid * 16 * 512;
  const int qr = sampled[b * 256 + lt * 16 + lo];

  for (int hh = 0; hh < 2; ++hh) {
    const int h = wid + hh * 4;
    float rm[8], ri[8];
#pragma unroll
    for (int v = 0; v < 8; ++v) {
      int m  = v + 8 * kh;
      int qm = sampled[b * 256 + lt * 16 + m];
      size_t o = (size_t)(b * 8 + h) * 512 + qm;
      rm[v] = rmax[o];
      ri[v] = rinv[o];
    }
    const __bf16* qbase = qkv + (size_t)(b * 512 + qr) * 768 + h * 32;
    v16bf qa = load_a16(qbase + kh * 16, qbase + kh * 16 + 8);
    for (int kt = 0; kt < 32; ++kt) {
      const __bf16* kbase = qkv + (size_t)(b * 512 + kt * 16 + lo) * 768 + 256 + h * 32 + kh * 16;
      v16bf kb = *(const v16bf*)kbase;
      v8f s = {};
      s = wmma_bf16(qa, kb, s);
#pragma unroll
      for (int v = 0; v < 8; ++v) {
        int m = v + 8 * kh;
        float p = __expf(s[v] - rm[v]) * ri[v];
        Pw[m * 512 + kt * 16 + lo] = (__bf16)p;
      }
    }
    __syncthreads();
#pragma unroll
    for (int nt = 0; nt < 2; ++nt) {
      v8f acc = {};
      for (int ks = 0; ks < 16; ++ks) {
        int k0 = ks * 32;
        const __bf16* pl = Pw + lo * 512 + k0;
        v16bf pa = load_a16(pl + kh * 16, pl + kh * 16 + 8);
        const __bf16* vb = vT + (size_t)((b * 8 + h) * 32 + nt * 16 + lo) * 512 + k0 + kh * 16;
        v16bf bb = *(const v16bf*)vb;
        acc = wmma_bf16(pa, bb, acc);
      }
#pragma unroll
      for (int v = 0; v < 8; ++v) {
        int m = v + 8 * kh;
        Obf[(size_t)(b * 256 + lt * 16 + m) * 256 + h * 32 + nt * 16 + lo] = (__bf16)acc[v];
      }
    }
    __syncthreads();
  }
}

// ---------------- residual + layernorm kernels ----------------
__global__ void k_add_ln_hm(const float* __restrict__ src, const float* __restrict__ hpre,
                            const int* __restrict__ sampled, const float* __restrict__ g,
                            const float* __restrict__ bt, float* __restrict__ hm,
                            __bf16* __restrict__ hmbf) {
  __shared__ float red[256];
  int row = blockIdx.x, d = threadIdx.x;
  int b = row >> 8, li = row & 255;
  int q = sampled[b * 256 + li];
  float v = src[((size_t)(b * 513 + 1 + q)) * 256 + d] + hpre[(size_t)row * 256 + d];
  float mean = blk_sum256(v, red) * (1.f / 256.f);
  float c = v - mean;
  float var = blk_sum256(c * c, red) * (1.f / 256.f);
  float o = c * rsqrtf(var + 1e-5f) * g[d] + bt[d];
  hm[(size_t)row * 256 + d] = o;
  hmbf[(size_t)row * 256 + d] = (__bf16)o;
}

__global__ void k_add_ln_intra(const float* __restrict__ hm, const float* __restrict__ f2,
                               const float* __restrict__ g, const float* __restrict__ bt,
                               float* __restrict__ out, __bf16* __restrict__ intrabf) {
  __shared__ float red[256];
  int row = blockIdx.x, d = threadIdx.x;
  int b = row >> 8, li = row & 255;
  float v = hm[(size_t)row * 256 + d] + f2[(size_t)row * 256 + d];
  float mean = blk_sum256(v, red) * (1.f / 256.f);
  float c = v - mean;
  float var = blk_sum256(c * c, red) * (1.f / 256.f);
  float o = c * rsqrtf(var + 1e-5f) * g[d] + bt[d];
  out[((size_t)(b * 257 + 1 + li)) * 256 + d] = o;   // src_out[:,1:] = intra
  intrabf[(size_t)row * 256 + d] = (__bf16)o;
}

// ---------------- per-batch token path (tiny; pure VALU) ----------------
__global__ void k_token(const float* __restrict__ src, const float* __restrict__ kv,
                        const float* __restrict__ attn_in_w, const float* __restrict__ attn_in_b,
                        const float* __restrict__ attn_out_w, const float* __restrict__ attn_out_b,
                        const float* __restrict__ lin1_w, const float* __restrict__ lin1_b,
                        const float* __restrict__ lin2_w, const float* __restrict__ lin2_b,
                        const float* __restrict__ n1g, const float* __restrict__ n1b,
                        const float* __restrict__ n2g, const float* __restrict__ n2b,
                        float* __restrict__ out) {
  __shared__ float tok[256], tq[256], att[256], so[256], t2s[256], red[256], ffh[2048];
  int b = blockIdx.x, d = threadIdx.x;
  tok[d] = src[((size_t)(b * 513)) * 256 + d];
  __syncthreads();
  { // tq = (token @ wq^T + bq) / 8
    float a = 0.f;
    const float* w = attn_in_w + (size_t)d * 256;
    for (int k = 0; k < 256; ++k) a += tok[k] * w[k];
    tq[d] = (a + attn_in_b[d]) * 0.125f;
  }
  __syncthreads();
  for (int h = 0; h < 4; ++h) {
    float s = 0.f;
    const float* kp = kv + ((size_t)(b * 256 + d)) * 512 + h * 64;
    for (int j = 0; j < 64; ++j) s += tq[h * 64 + j] * kp[j];
    red[d] = s; __syncthreads();
#pragma unroll
    for (int st = 128; st > 0; st >>= 1) {
      if (d < st) red[d] = fmaxf(red[d], red[d + st]);
      __syncthreads();
    }
    float mx = red[0]; __syncthreads();
    float e = __expf(s - mx);
    float ssum = blk_sum256(e, red);
    att[d] = e / ssum;
    __syncthreads();
    if (d < 64) {
      float o = 0.f;
      for (int t = 0; t < 256; ++t) o += att[t] * kv[((size_t)(b * 256 + t)) * 512 + 256 + h * 64 + d];
      so[h * 64 + d] = o;
    }
    __syncthreads();
  }
  float ss = 0.f;
  { const float* w = attn_out_w + (size_t)d * 256;
    for (int k = 0; k < 256; ++k) ss += so[k] * w[k]; }
  float t1 = tok[d] + ss + attn_out_b[d];
  float mean = blk_sum256(t1, red) * (1.f / 256.f);
  float c = t1 - mean;
  float var = blk_sum256(c * c, red) * (1.f / 256.f);
  float tv2 = c * rsqrtf(var + 1e-5f) * n1g[d] + n1b[d];
  t2s[d] = tv2;
  __syncthreads();
  for (int r = 0; r < 8; ++r) {
    int idx = r * 256 + d;
    float a = lin1_b[idx];
    const float* w = lin1_w + (size_t)idx * 256;
    for (int k = 0; k < 256; ++k) a += t2s[k] * w[k];
    ffh[idx] = fmaxf(a, 0.f);
  }
  __syncthreads();
  float f = lin2_b[d];
  { const float* w = lin2_w + (size_t)d * 2048;
    for (int j = 0; j < 2048; ++j) f += ffh[j] * w[j]; }
  float t3 = tv2 + f;
  mean = blk_sum256(t3, red) * (1.f / 256.f);
  c = t3 - mean;
  var = blk_sum256(c * c, red) * (1.f / 256.f);
  float to = c * rsqrtf(var + 1e-5f) * n2g[d] + n2b[d];
  out[((size_t)(b * 257)) * 256 + d] = to;                 // src_out[:,0]
  int g = b >> 3, i = b & 7;
  out[OUT_GROUPED + (size_t)(i * 4 + g) * 256 + d] = to;   // grouped[i,g,:]
}

__global__ void k_gather_idx(const int* __restrict__ src_idx, const int* __restrict__ sampled,
                             int* __restrict__ outi) {
  int i = blockIdx.x * 256 + threadIdx.x;    // 32*256
  int b = i >> 8;
  outi[i] = src_idx[b * 512 + sampled[i]];
}

// ---------------------------------------------------------------------------
extern "C" void kernel_launch(void* const* d_in, const int* in_sizes, int n_in,
                              void* d_out, int out_size, void* d_ws, size_t ws_size,
                              hipStream_t stream) {
  (void)in_sizes; (void)n_in; (void)out_size; (void)ws_size;
  const float* src        = (const float*)d_in[0];
  const int*   src_idx    = (const int*)  d_in[1];
  const float* mix_in_w   = (const float*)d_in[2];
  const float* mix_in_b   = (const float*)d_in[3];
  const float* mix_out_w  = (const float*)d_in[4];
  const float* mix_out_b  = (const float*)d_in[5];
  const float* mnorm_g    = (const float*)d_in[6];
  const float* mnorm_b    = (const float*)d_in[7];
  const float* mnormc_g   = (const float*)d_in[8];
  const float* mnormc_b   = (const float*)d_in[9];
  const float* ffn_w1     = (const float*)d_in[10];
  const float* ffn_b1     = (const float*)d_in[11];
  const float* ffn_w2     = (const float*)d_in[12];
  const float* ffn_b2     = (const float*)d_in[13];
  const float* attn_in_w  = (const float*)d_in[14];
  const float* attn_in_b  = (const float*)d_in[15];
  const float* attn_out_w = (const float*)d_in[16];
  const float* attn_out_b = (const float*)d_in[17];
  const float* lin1_w     = (const float*)d_in[18];
  const float* lin1_b     = (const float*)d_in[19];
  const float* lin2_w     = (const float*)d_in[20];
  const float* lin2_b     = (const float*)d_in[21];
  const float* n1_g       = (const float*)d_in[22];
  const float* n1_b       = (const float*)d_in[23];
  const float* n2_g       = (const float*)d_in[24];
  const float* n2_b       = (const float*)d_in[25];

  char* ws = (char*)d_ws;
  __bf16* mixw_bf  = (__bf16*)(ws + WS_MIXW);
  __bf16* outw_bf  = (__bf16*)(ws + WS_OUTW);
  __bf16* ff1w_bf  = (__bf16*)(ws + WS_FF1W);
  __bf16* ff2w_bf  = (__bf16*)(ws + WS_FF2W);
  __bf16* attw_bf  = (__bf16*)(ws + WS_ATTW);
  float*  mixb_s   = (float*) (ws + WS_MIXB);
  __bf16* x_bf     = (__bf16*)(ws + WS_XBF);
  __bf16* O_bf     = (__bf16*)(ws + WS_XBF);    // alias: x_bf dead after qkv GEMM
  __bf16* qkv_bf   = (__bf16*)(ws + WS_QKV);
  __bf16* vT       = (__bf16*)(ws + WS_VT);
  float*  rmax     = (float*) (ws + WS_RMAX);
  float*  rinv     = (float*) (ws + WS_RINV);
  float*  wgt      = (float*) (ws + WS_WGT);
  int*    sampled  = (int*)   (ws + WS_SAMP);
  float*  hpre     = (float*) (ws + WS_HPRE);   // also ffn2 output
  float*  hm       = (float*) (ws + WS_HM);
  float*  kvbuf    = (float*) (ws + WS_KV);
  __bf16* hm_bf    = (__bf16*)(ws + WS_HMBF);   // aliases inside dead qkv region
  __bf16* ff1o_bf  = (__bf16*)(ws + WS_FF1O);
  __bf16* intra_bf = (__bf16*)(ws + WS_INTRB);
  float*  out      = (float*)d_out;

  // weight / input conversion (q weight+bias pre-scaled by 1/sqrt(dh))
  k_cvt_scale<<<256, 256, 0, stream>>>(mix_in_w,          mixw_bf,          65536, 0.17677669529663687f);
  k_cvt_scale<<<512, 256, 0, stream>>>(mix_in_w + 65536,  mixw_bf + 65536, 131072, 1.f);
  k_cvt_scale<<<256, 256, 0, stream>>>(mix_out_w, outw_bf,  65536, 1.f);
  k_cvt_scale<<<512, 256, 0, stream>>>(ffn_w1,    ff1w_bf, 131072, 1.f);
  k_cvt_scale<<<512, 256, 0, stream>>>(ffn_w2,    ff2w_bf, 131072, 1.f);
  k_cvt_scale<<<768, 256, 0, stream>>>(attn_in_w, attw_bf, 196608, 1.f);
  k_scale_qbias<<<3, 256, 0, stream>>>(mix_in_b, mixb_s);
  k_cvt_x<<<16384, 256, 0, stream>>>(src, x_bf);
  k_zero_f32<<<64, 256, 0, stream>>>(wgt, 16384);

  // qkv = x @ mix_in_w^T + b   (bf16 out, q pre-scaled)
  k_gemm_bf16<<<dim3(1024, 12), 32, 0, stream>>>(x_bf, mixw_bf, mixb_s, (void*)qkv_bf, 16384, 768, 256, 2);
  k_make_vT<<<16384, 256, 0, stream>>>(qkv_bf, vT);

  // flash-style attention stats + head-max/query-sum weights
  k_attn_pass1<<<1024, 256, 0, stream>>>(qkv_bf, rmax, rinv, wgt);
  k_topk_rank<<<32, 512, 0, stream>>>(wgt, sampled);
  k_attn_pass2<<<512, 128, 0, stream>>>(qkv_bf, vT, rmax, rinv, sampled, O_bf);

  // src2 = O @ mix_out_w^T + b ; h = x_sel + src2 ; hm = LN(h)
  k_gemm_bf16<<<dim3(512, 4), 32, 0, stream>>>(O_bf, outw_bf, mix_out_b, (void*)hpre, 8192, 256, 256, 0);
  k_add_ln_hm<<<8192, 256, 0, stream>>>(src, hpre, sampled, mnorm_g, mnorm_b, hm, hm_bf);

  // hm = hm + relu(hm@W1^T+b1)@W2^T + b2 ; intra = LN(hm)
  k_gemm_bf16<<<dim3(512, 8), 32, 0, stream>>>(hm_bf,   ff1w_bf, ffn_b1, (void*)ff1o_bf, 8192, 512, 256, 3);
  k_gemm_bf16<<<dim3(512, 4), 32, 0, stream>>>(ff1o_bf, ff2w_bf, ffn_b2, (void*)hpre,    8192, 256, 512, 0);
  k_add_ln_intra<<<8192, 256, 0, stream>>>(hm, hpre, mnormc_g, mnormc_b, out, intra_bf);

  // kv = intra @ [wk;wv]^T + [bk;bv]
  k_gemm_bf16<<<dim3(512, 8), 32, 0, stream>>>(intra_bf, attw_bf + 65536, attn_in_b + 256,
                                               (void*)kvbuf, 8192, 512, 256, 0);

  // token attention + FFN + norms ; writes src_out[:,0] and grouped
  k_token<<<32, 256, 0, stream>>>(src, kvbuf, attn_in_w, attn_in_b, attn_out_w, attn_out_b,
                                  lin1_w, lin1_b, lin2_w, lin2_b, n1_g, n1_b, n2_g, n2_b, out);
  k_gather_idx<<<32, 256, 0, stream>>>(src_idx, sampled, (int*)d_out + OUT_IDX);
}
